// HailNet_42975442763785
// MI455X (gfx1250) — matrix operand order repaired and checked by reading, not compiled, not run on hardware
//
#include <hip/hip_runtime.h>
#include <hip/hip_bf16.h>

// ---------- types for WMMA ----------
typedef __attribute__((ext_vector_type(8)))  __bf16 v8bf;
typedef __attribute__((ext_vector_type(16))) __bf16 v16bf;
typedef __attribute__((ext_vector_type(8)))  float  v8f;

#define NFLAT 65536
#define BT    48
#define EMB   256
#define HID   256
#define TSEQ  12
#define KSPLIT 64          // K-chunks for the big GEMM (65536/64 = 1024 K each)

// round-to-nearest-even f32 -> bf16
static __device__ __forceinline__ unsigned short f2bf(float f) {
    unsigned int u = __float_as_uint(f);
    u += 0x7FFFu + ((u >> 16) & 1u);
    return (unsigned short)(u >> 16);
}
static __device__ __forceinline__ float sigmf(float x) {
    return 1.0f / (1.0f + __expf(-x));
}
static __device__ __forceinline__ v16bf joinbf(v8bf lo, v8bf hi) {
    return __builtin_shufflevector(lo, hi, 0,1,2,3,4,5,6,7,
                                           8,9,10,11,12,13,14,15);
}

// ---------- utility kernels ----------
__global__ void zero_f32(float* __restrict__ p, int n) {
    int i = blockIdx.x * blockDim.x + threadIdx.x;
    if (i < n) p[i] = 0.0f;
}

__global__ void cvt_f32_bf16(const float* __restrict__ src,
                             unsigned short* __restrict__ dst, int n) {
    int i = blockIdx.x * blockDim.x + threadIdx.x;
    if (i < n) dst[i] = f2bf(src[i]);
}

// t2 = sigmoid(acc + bias) -> bf16, acc is [BT, EMB] fp32 partial sums
__global__ void finalize_t2(const float* __restrict__ acc,
                            const float* __restrict__ bias,
                            unsigned short* __restrict__ dst, int n) {
    int i = blockIdx.x * blockDim.x + threadIdx.x;
    if (i < n) dst[i] = f2bf(sigmf(acc[i] + bias[i & (EMB - 1)]));
}

// ---------- SpMM: h1[b, r] += xf[b, c] for every COO edge (r, c) ----------
__global__ void spmm_edges(const float* __restrict__ xf,
                           const int* __restrict__ rows,
                           const int* __restrict__ cols,
                           float* __restrict__ h1, int nnz) {
    int e = blockIdx.x * blockDim.x + threadIdx.x;
    if (e >= nnz) return;
    int b = blockIdx.y;
    int r = rows[e], c = cols[e];
    atomicAdd(&h1[(size_t)b * NFLAT + r], xf[(size_t)b * NFLAT + c]);
}

// ---------- bf16 WMMA GEMM:  out[M,N] = act(A[M,K] @ Bm[N,K]^T [+ bias]) ----
// One wave per (16x16 tile, K-chunk). grid = (N/16, M/16, nchunks), block = 32.
// Requires (kend - k0) to be a positive multiple of 64.
// mode 0: full-K, sigmoid -> bf16 out
// mode 1: full-K, +bias    -> f32 out
// mode 2: split-K partial  -> fp32 atomicAdd into outF (bias applied later)
__global__ __launch_bounds__(32)
void gemm_bf16_wmma(const unsigned short* __restrict__ A, int lda,
                    const unsigned short* __restrict__ Bm, int ldb,
                    const float* __restrict__ bias,
                    unsigned short* __restrict__ outB,
                    float* __restrict__ outF,
                    int K, int kchunk, int ldo, int mode) {
    const int lane = threadIdx.x & 31;
    const int lm   = lane & 15;
    const int hi   = lane >> 4;   // 0: lanes 0-15, 1: lanes 16-31
    const int mtile = blockIdx.y, ntile = blockIdx.x;
    const int k0   = blockIdx.z * kchunk;
    const int kend = (k0 + kchunk < K) ? (k0 + kchunk) : K;

    // A 16x32 layout (matches B): hi=0 K=0..15, hi=1 K=16..31
    const unsigned short* pa = A  + (size_t)(mtile * 16 + lm) * lda + hi * 16;
    // B 32x16 layout (same contiguous k as A): lane = column; hi=0 K=0..15, hi=1 K=16..31
    const unsigned short* pb = Bm + (size_t)(ntile * 16 + lm) * ldb + hi * 16;

    v8f acc = {};
    // ping-pong double buffering, unrolled by 2: no register-rotation movs,
    // next block's loads are in flight while the WMMA consumes the previous.
    int k = k0;
    v8bf  a00 = *(const v8bf*)(pa + k);
    v8bf  a01 = *(const v8bf*)(pa + k + 8);
    v16bf b0  = *(const v16bf*)(pb + k);
    for (; k + 64 < kend; k += 64) {
        v8bf  a10 = *(const v8bf*)(pa + k + 32);
        v8bf  a11 = *(const v8bf*)(pa + k + 40);
        v16bf b1  = *(const v16bf*)(pb + k + 32);
        if (k + 512 < kend) {                     // -> global_prefetch_b8
            __builtin_prefetch(pa + k + 512, 0, 1);
            __builtin_prefetch(pb + k + 512, 0, 1);
        }
        acc = __builtin_amdgcn_wmma_f32_16x16x32_bf16(
                  false, joinbf(a00, a01), false, b0, (short)0, acc, false, false);
        a00 = *(const v8bf*)(pa + k + 64);
        a01 = *(const v8bf*)(pa + k + 72);
        b0  = *(const v16bf*)(pb + k + 64);
        acc = __builtin_amdgcn_wmma_f32_16x16x32_bf16(
                  false, joinbf(a10, a11), false, b1, (short)0, acc, false, false);
    }
    {   // final two 32-K blocks (block count is even)
        v8bf  a10 = *(const v8bf*)(pa + k + 32);
        v8bf  a11 = *(const v8bf*)(pa + k + 40);
        v16bf b1  = *(const v16bf*)(pb + k + 32);
        acc = __builtin_amdgcn_wmma_f32_16x16x32_bf16(
                  false, joinbf(a00, a01), false, b0, (short)0, acc, false, false);
        acc = __builtin_amdgcn_wmma_f32_16x16x32_bf16(
                  false, joinbf(a10, a11), false, b1, (short)0, acc, false, false);
    }

    // C/D layout: element r -> row mtile*16 + hi*8 + r, col ntile*16 + lm
    const int col = ntile * 16 + lm;
#pragma unroll
    for (int r = 0; r < 8; ++r) {
        const int row = mtile * 16 + hi * 8 + r;
        if (mode == 2) {
            atomicAdd(&outF[(size_t)row * ldo + col], acc[r]);
        } else if (mode == 0) {
            outB[(size_t)row * ldo + col] = f2bf(sigmf(acc[r] + bias[col]));
        } else {
            outF[(size_t)row * ldo + col] = acc[r] + bias[col];
        }
    }
}

// ---------- persistent GRU + output MLP (tiny, sequential) ----------
__global__ __launch_bounds__(1024)
void gru_tail(const float* __restrict__ h0,
              const float* __restrict__ xproj,   // [48, 768], row = b*T + t
              const float* __restrict__ Whh, const float* __restrict__ bhh,
              const float* __restrict__ Wf0, const float* __restrict__ bf0,
              const float* __restrict__ Wf1, const float* __restrict__ bf1,
              const float* __restrict__ Wf2, const float* __restrict__ bf2,
              float* __restrict__ out) {
    __shared__ float h[4 * HID];
    __shared__ float gh[4 * 3 * HID];
    __shared__ float o0[4 * 16];
    __shared__ float o1[4 * 16];
    const int tid = threadIdx.x;

    h[tid] = h0[tid];
    __syncthreads();

    for (int t = 0; t < TSEQ; ++t) {
#pragma unroll
        for (int ii = 0; ii < 3; ++ii) {
            const int idx = tid * 3 + ii;           // 0..3071
            const int b = idx / 768, j = idx - b * 768;
            const float* hb = &h[b * HID];
            const float* w  = &Whh[(size_t)j * HID];
            float s = bhh[j];
#pragma unroll 4
            for (int kk = 0; kk < HID; ++kk) s += hb[kk] * w[kk];
            gh[idx] = s;
        }
        __syncthreads();
        {   // gate math: torch order (r, z, n)
            const int b = tid >> 8, j = tid & 255;
            const float* xp = &xproj[(size_t)(b * TSEQ + t) * 768];
            float xr = xp[j], xz = xp[256 + j], xn = xp[512 + j];
            float hr = gh[b * 768 + j], hz = gh[b * 768 + 256 + j],
                  hn = gh[b * 768 + 512 + j];
            float r = sigmf(xr + hr);
            float z = sigmf(xz + hz);
            float nw = tanhf(xn + r * hn);
            h[b * HID + j] = (1.0f - z) * nw + z * h[b * HID + j];
        }
        __syncthreads();
    }

    if (tid < 64) {
        const int b = tid >> 4, j = tid & 15;
        float s = bf0[j];
        for (int kk = 0; kk < HID; ++kk) s += h[b * HID + kk] * Wf0[j * HID + kk];
        o0[b * 16 + j] = sigmf(s);
    }
    __syncthreads();
    if (tid < 64) {
        const int b = tid >> 4, j = tid & 15;
        float s = bf1[j];
        for (int kk = 0; kk < 16; ++kk) s += o0[b * 16 + kk] * Wf1[j * 16 + kk];
        o1[b * 16 + j] = sigmf(s);
    }
    __syncthreads();
    if (tid < 4) {
        float s = bf2[0];
        for (int kk = 0; kk < 16; ++kk) s += o1[tid * 16 + kk] * Wf2[kk];
        out[tid] = sigmf(s);
    }
}

extern "C" void kernel_launch(void* const* d_in, const int* in_sizes, int n_in,
                              void* d_out, int out_size, void* d_ws, size_t ws_size,
                              hipStream_t stream) {
    const float* x    = (const float*)d_in[0];
    const float* h0   = (const float*)d_in[1];
    const int*   rows = (const int*)d_in[2];
    const int*   cols = (const int*)d_in[3];
    const float* Wemb = (const float*)d_in[4];
    const float* bemb = (const float*)d_in[5];
    const float* Wl1  = (const float*)d_in[6];
    const float* bl1  = (const float*)d_in[7];
    const float* Wih  = (const float*)d_in[8];
    const float* Whh  = (const float*)d_in[9];
    const float* bih  = (const float*)d_in[10];
    const float* bhh  = (const float*)d_in[11];
    const float* Wf0  = (const float*)d_in[12];
    const float* bf0  = (const float*)d_in[13];
    const float* Wf1  = (const float*)d_in[14];
    const float* bf1  = (const float*)d_in[15];
    const float* Wf2  = (const float*)d_in[16];
    const float* bf2  = (const float*)d_in[17];
    float* out = (float*)d_out;
    const int nnz = in_sizes[2];

    // workspace carve-out (256B aligned)
    char* ws = (char*)d_ws;
    size_t off = 0;
    auto carve = [&](size_t bytes) {
        void* p = ws + off;
        off = (off + bytes + 255) & ~(size_t)255;
        return p;
    };
    float*          h1f   = (float*)         carve((size_t)BT * NFLAT * 4);
    unsigned short* h1b   = (unsigned short*)carve((size_t)BT * NFLAT * 2);
    unsigned short* wembb = (unsigned short*)carve((size_t)EMB * NFLAT * 2);
    unsigned short* wl1b  = (unsigned short*)carve((size_t)EMB * EMB * 2);
    unsigned short* wihb  = (unsigned short*)carve((size_t)3 * HID * EMB * 2);
    float*          t2f   = (float*)         carve((size_t)BT * EMB * 4);
    unsigned short* t2b   = (unsigned short*)carve((size_t)BT * EMB * 2);
    unsigned short* t4b   = (unsigned short*)carve((size_t)BT * EMB * 2);
    float*          xproj = (float*)         carve((size_t)BT * 3 * HID * 4);

    // 1) h1 = A @ xf.T (fp32 atomics over COO edges)
    zero_f32<<<(BT * NFLAT) / 256, 256, 0, stream>>>(h1f, BT * NFLAT);
    spmm_edges<<<dim3((nnz + 255) / 256, BT), 256, 0, stream>>>(x, rows, cols, h1f, nnz);

    // 2) fp32 -> bf16 conversions for WMMA operands; zero split-K accumulator
    cvt_f32_bf16<<<(BT * NFLAT + 255) / 256, 256, 0, stream>>>(h1f, h1b, BT * NFLAT);
    cvt_f32_bf16<<<(EMB * NFLAT + 255) / 256, 256, 0, stream>>>(Wemb, wembb, EMB * NFLAT);
    cvt_f32_bf16<<<(EMB * EMB + 255) / 256, 256, 0, stream>>>(Wl1, wl1b, EMB * EMB);
    cvt_f32_bf16<<<(3 * HID * EMB + 255) / 256, 256, 0, stream>>>(Wih, wihb, 3 * HID * EMB);
    zero_f32<<<(BT * EMB + 255) / 256, 256, 0, stream>>>(t2f, BT * EMB);

    // 3) split-K big GEMM: t2f += h1 @ Wemb^T   [48,65536] x [65536,256]
    gemm_bf16_wmma<<<dim3(EMB / 16, BT / 16, KSPLIT), 32, 0, stream>>>(
        h1b, NFLAT, wembb, NFLAT, nullptr, nullptr, t2f,
        NFLAT, NFLAT / KSPLIT, EMB, 2);
    finalize_t2<<<(BT * EMB + 255) / 256, 256, 0, stream>>>(t2f, bemb, t2b, BT * EMB);

    // 4) t4 = sigmoid(t2 @ Wl1^T + bl1)     [48,256] x [256,256]
    gemm_bf16_wmma<<<dim3(EMB / 16, BT / 16, 1), 32, 0, stream>>>(
        t2b, EMB, wl1b, EMB, bl1, t4b, nullptr, EMB, EMB, EMB, 0);

    // 5) x_proj = t4 @ Wih^T + bih          [48,256] x [256,768]
    gemm_bf16_wmma<<<dim3(3 * HID / 16, BT / 16, 1), 32, 0, stream>>>(
        t4b, EMB, wihb, EMB, bih, nullptr, xproj, EMB, EMB, 3 * HID, 1);

    // 6) GRU over T=12 + output MLP -> out[4]
    gru_tail<<<1, 1024, 0, stream>>>(h0, xproj, Whh, bhh,
                                     Wf0, bf0, Wf1, bf1, Wf2, bf2, out);
}
